// CSDivergenceLoss_50079318672069
// MI455X (gfx1250) — matrix operand order repaired and checked by reading, not compiled, not run on hardware
//
#include <hip/hip_runtime.h>
#include <hip/hip_bf16.h>
#include <math.h>

typedef __attribute__((ext_vector_type(16))) _Float16 v16h;
typedef __attribute__((ext_vector_type(8)))  _Float16 v8h;
typedef __attribute__((ext_vector_type(8)))  float    v8f;

#define BS      32
#define KP      1000
#define KG      100
#define NC      80
#define KP_PAD  1008      // 63 * 16
#define K_PAD   96        // 3 * 32 (WMMA K steps)
#define NTIL    63
#define SSTR    8         // stat record (floats): {m0/v0, m1/v1, 1/v0, 1/v1, qpref, l2v, 0, 0}
#define LOG2E_F    1.4426950408889634f   // log2(e)
#define LN2_F      0.6931471805599453f   // ln(2)
#define LOG2_2PI_F 2.6514961294723187f   // log2(2*pi)

__device__ __forceinline__ float fastRcp(float x)  { return __builtin_amdgcn_rcpf(x); }
__device__ __forceinline__ float fastLog2(float x) { return __builtin_amdgcn_logf(x); }   // v_log_f32
__device__ __forceinline__ float fastExp2(float x) { return __builtin_amdgcn_exp2f(x); }  // v_exp_f32

__device__ __forceinline__ float waveReduceSum(float v) {
  #pragma unroll
  for (int off = 16; off >= 1; off >>= 1) v += __shfl_xor(v, off, 32);
  return v;
}

// ---------------------------------------------------------------------------
// init: zero the pq/qq atomic accumulators
// ---------------------------------------------------------------------------
__global__ void init_acc(float* __restrict__ pq, float* __restrict__ qq) {
  int t = threadIdx.x;
  if (t < BS) { pq[t] = 0.f; qq[t] = 0.f; }
}

// Stat record in log2 domain: s4 = 0.5*log2e * sum(m^2/v); s5 = 0.5*sum(log2 v)
__device__ __forceinline__ void write_stat(float* __restrict__ s,
                                           float m0, float m1, float v0, float v1) {
  float c0 = fastRcp(v0), c1 = fastRcp(v1);
  s[0] = m0 * c0; s[1] = m1 * c1;
  s[2] = c0;      s[3] = c1;
  s[4] = 0.5f * LOG2E_F * (m0 * m0 * c0 + m1 * m1 * c1);
  s[5] = 0.5f * (fastLog2(v0) + fastLog2(v1));
  s[6] = 0.f;     s[7] = 0.f;
}

// exp(delta_ij) computed fully in log2 domain: 2x v_rcp + 2x v_log + 1x v_exp.
__device__ __forceinline__ float pair_w(const float* __restrict__ si,
                                        const float* __restrict__ sj) {
  float s0 = si[2] + sj[2], s1 = si[3] + sj[3];
  float n0 = si[0] + sj[0], n1 = si[1] + sj[1];
  float q  = n0 * n0 * fastRcp(s0) + n1 * n1 * fastRcp(s1);
  float e2 = 0.5f * LOG2E_F * q - si[4] - sj[4]
           - 0.5f * (fastLog2(s0) + fastLog2(s1))
           - si[5] - sj[5] - LOG2_2PI_F;
  return fastExp2(e2);
}

// ---------------------------------------------------------------------------
// pred preprocess: one wave per row. softmax(80)*sigmoid -> alpha
// (f32 class-major for pq, f16 row-major for WMMA-A and B, f16 transposed
// not read by qq_wmma), plus 8-float Gaussian row stats. Rows >= KP are padding.
// ---------------------------------------------------------------------------
__global__ void pred_prep(const float* __restrict__ pred_bboxes,
                          const float* __restrict__ pred_labels,
                          float*    __restrict__ palphaC,   // [BS][NC][KP]
                          _Float16* __restrict__ palphaH,   // [BS][KP_PAD][K_PAD]
                          _Float16* __restrict__ palphaT,   // [BS][K_PAD][KP_PAD]
                          float*    __restrict__ pstat) {   // [BS][KP_PAD][SSTR]
  int row = blockIdx.x;                   // 0 .. BS*KP_PAD-1
  int b = row / KP_PAD, j = row % KP_PAD;
  int t = threadIdx.x;                    // 0..31
  _Float16* hrow = palphaH + (size_t)(b * KP_PAD + j) * K_PAD;
  _Float16* tcol = palphaT + (size_t)b * K_PAD * KP_PAD + j;  // + k*KP_PAD
  float*    srow = pstat   + (size_t)(b * KP_PAD + j) * SSTR;
  if (j >= KP) {
    for (int k = t; k < K_PAD; k += 32) {
      hrow[k] = (_Float16)0.f;
      tcol[(size_t)k * KP_PAD] = (_Float16)0.f;
    }
    if (t == 0) write_stat(srow, 0.f, 0.f, 1.f, 1.f);
    return;
  }
  const float* lrow = pred_labels + (size_t)(b * KP + j) * (NC + 1);
  float x0 = lrow[t], x1 = lrow[t + 32];
  float x2 = (t < 16) ? lrow[t + 64] : -1e30f;
  float mx = fmaxf(fmaxf(x0, x1), x2);
  #pragma unroll
  for (int off = 16; off >= 1; off >>= 1) mx = fmaxf(mx, __shfl_xor(mx, off, 32));
  float e0 = fastExp2(LOG2E_F * (x0 - mx)), e1 = fastExp2(LOG2E_F * (x1 - mx));
  float e2 = (t < 16) ? fastExp2(LOG2E_F * (x2 - mx)) : 0.f;
  float den = waveReduceSum(e0 + e1 + e2);
  float sig = fastRcp(1.f + fastExp2(-LOG2E_F * lrow[NC]));
  float scale = sig * fastRcp(den);
  float* acol = palphaC + (size_t)b * NC * KP + j;            // + k*KP
  float a0 = e0 * scale, a1 = e1 * scale;
  acol[(size_t)t * KP]        = a0;
  acol[(size_t)(t + 32) * KP] = a1;
  hrow[t]      = (_Float16)a0;  tcol[(size_t)t * KP_PAD]        = (_Float16)a0;
  hrow[t + 32] = (_Float16)a1;  tcol[(size_t)(t + 32) * KP_PAD] = (_Float16)a1;
  if (t < 16) {
    float a2 = e2 * scale;
    acol[(size_t)(t + 64) * KP] = a2;
    hrow[t + 64] = (_Float16)a2;  tcol[(size_t)(t + 64) * KP_PAD] = (_Float16)a2;
    hrow[t + 80] = (_Float16)0.f; tcol[(size_t)(t + 80) * KP_PAD] = (_Float16)0.f;  // K pad
  }
  if (t == 0) {
    const float* bb = pred_bboxes + (size_t)(b * KP + j) * 4;
    float hw = bb[2] * 0.5f, hh = bb[3] * 0.5f;
    write_stat(srow, bb[0], bb[1], hw * hw, hh * hh);
  }
}

// ---------------------------------------------------------------------------
// gt stats
// ---------------------------------------------------------------------------
__global__ void gt_prep(const float* __restrict__ gt_bboxes,
                        float* __restrict__ gstat, int n) {
  int r = blockIdx.x * blockDim.x + threadIdx.x;
  if (r >= n) return;
  const float* bb = gt_bboxes + (size_t)r * 4;
  float hw = bb[2] * 0.5f, hh = bb[3] * 0.5f;
  write_stat(gstat + (size_t)r * SSTR, bb[0], bb[1], hw * hw, hh * hh);
}

// ---------------------------------------------------------------------------
// pq: one block per (b, gt i); w[i,j] = pred_alpha[j, label_i] (one-hot gemm)
// alpha stored class-major -> coalesced b32 reads over j.
// ---------------------------------------------------------------------------
__global__ void pq_kernel(const float* __restrict__ palphaC,
                          const float* __restrict__ pstat,
                          const float* __restrict__ gstat,
                          const int*   __restrict__ gt_labels,
                          float* __restrict__ pq) {
  int blk = blockIdx.x;
  int b = blk / KG, i = blk % KG;
  const float* sg = gstat + (size_t)(b * KG + i) * SSTR;
  float si[6];
  #pragma unroll
  for (int q = 0; q < 6; ++q) si[q] = sg[q];
  int lab = gt_labels[b * KG + i];
  const float* wrow = palphaC + ((size_t)b * NC + lab) * KP;
  float part = 0.f;
  for (int j = threadIdx.x; j < KP; j += blockDim.x) {
    const float* sj = pstat + (size_t)(b * KP_PAD + j) * SSTR;
    part += wrow[j] * pair_w(si, sj);
  }
  __shared__ float red[256];
  red[threadIdx.x] = part;
  __syncthreads();
  #pragma unroll
  for (int s = 128; s > 0; s >>= 1) {
    if (threadIdx.x < s) red[threadIdx.x] += red[threadIdx.x + s];
    __syncthreads();
  }
  if (threadIdx.x == 0) atomicAdd(&pq[b], red[0]);
}

// ---------------------------------------------------------------------------
// pp: one block per image; w = (label_i == label_j)
// ---------------------------------------------------------------------------
__global__ void pp_kernel(const float* __restrict__ gstat,
                          const int*   __restrict__ gt_labels,
                          float* __restrict__ pp) {
  int b = blockIdx.x;
  float part = 0.f;
  for (int p = threadIdx.x; p < KG * KG; p += blockDim.x) {
    int i = p / KG, j = p % KG;
    if (gt_labels[b * KG + i] == gt_labels[b * KG + j]) {
      part += pair_w(gstat + (size_t)(b * KG + i) * SSTR,
                     gstat + (size_t)(b * KG + j) * SSTR);
    }
  }
  __shared__ float red[256];
  red[threadIdx.x] = part;
  __syncthreads();
  #pragma unroll
  for (int s = 128; s > 0; s >>= 1) {
    if (threadIdx.x < s) red[threadIdx.x] += red[threadIdx.x + s];
    __syncthreads();
  }
  if (threadIdx.x == 0) pp[b] = red[0];
}

// ---------------------------------------------------------------------------
// qq: fused WMMA tile GEMM (W = alpha alpha^T, f16 in / f32 acc) + pairwise
// exp2 weighting. 4 waves per block, one 16x16 tile per wave.
// All 12 fragment b128 loads issued up front; 3x v_wmma_f32_16x16x32_f16.
// ---------------------------------------------------------------------------
__global__ void __launch_bounds__(128)
qq_wmma(const _Float16* __restrict__ alphaH,
        const _Float16* __restrict__ alphaT,
        const float*    __restrict__ pstat,
        float* __restrict__ qq) {
  int lane = threadIdx.x & 31;
  int gtile = blockIdx.x * 4 + (threadIdx.x >> 5);
  int b  = gtile / (NTIL * NTIL);
  int tt = gtile % (NTIL * NTIL);
  int i0 = (tt / NTIL) * 16;
  int j0 = (tt % NTIL) * 16;
  int r = lane & 15, h = lane >> 4;

  // A: row r of i-tile; per K-step VGPRs 0-3 <- K=s+8h..+7, VGPRs 4-7 <- +16.
  const _Float16* Abase = alphaH + (size_t)(b * KP_PAD + i0 + r) * K_PAD + (h << 3);
  // B: column r of j-tile (alphaH row j0+r); same K order as A.
  const _Float16* Bbase = alphaH + (size_t)(b * KP_PAD + j0 + r) * K_PAD + (h << 3);

  // Preload all fragments (one big load clause; waits can be partial).
  v8h A[6], Bv[6];
  #pragma unroll
  for (int st = 0; st < 3; ++st) {
    A[2 * st]      = *(const v8h*)(Abase + 32 * st);
    A[2 * st + 1]  = *(const v8h*)(Abase + 32 * st + 16);
    Bv[2 * st]     = *(const v8h*)(Bbase + 32 * st);
    Bv[2 * st + 1] = *(const v8h*)(Bbase + 32 * st + 16);
  }

  v8f acc = {};
  #pragma unroll
  for (int st = 0; st < 3; ++st) {
    v16h a, bf;
    #pragma unroll
    for (int q = 0; q < 8; ++q) {
      a[q]  = A[2 * st][q];   a[8 + q]  = A[2 * st + 1][q];
      bf[q] = Bv[2 * st][q];  bf[8 + q] = Bv[2 * st + 1][q];
    }
    acc = __builtin_amdgcn_wmma_f32_16x16x32_f16(false, a, false, bf,
                                                 (short)0, acc, false, false);
  }

  // C/D layout: lane holds N = lane&15, rows M = vv + 8*(lane>>4)
  const float* sj = pstat + (size_t)(b * KP_PAD + j0 + (lane & 15)) * SSTR;
  float mjc0 = sj[0], mjc1 = sj[1];
  float cj0  = sj[2], cj1  = sj[3];
  float sj45 = sj[4] + sj[5] + LOG2_2PI_F;

  float part = 0.f;
  #pragma unroll
  for (int vv = 0; vv < 8; ++vv) {
    int m = vv + (h << 3);
    const float* si = pstat + (size_t)(b * KP_PAD + i0 + m) * SSTR;
    float s0 = si[2] + cj0, s1 = si[3] + cj1;
    float n0 = si[0] + mjc0, n1 = si[1] + mjc1;
    float q  = n0 * n0 * fastRcp(s0) + n1 * n1 * fastRcp(s1);
    float e2 = 0.5f * LOG2E_F * q - si[4] - si[5] - sj45
             - 0.5f * (fastLog2(s0) + fastLog2(s1));
    part += acc[vv] * fastExp2(e2);
  }
  part = waveReduceSum(part);
  if (lane == 0) atomicAdd(&qq[b], part);
}

// ---------------------------------------------------------------------------
// final: loss = -sum_b (2 log pq - log pp - log qq), ln = log2 * ln2
// ---------------------------------------------------------------------------
__global__ void final_kernel(const float* __restrict__ pq,
                             const float* __restrict__ pp,
                             const float* __restrict__ qq,
                             float* __restrict__ out) {
  int t = threadIdx.x;
  float v = (t < BS)
          ? LN2_F * (2.f * fastLog2(pq[t]) - fastLog2(pp[t]) - fastLog2(qq[t]))
          : 0.f;
  v = waveReduceSum(v);
  if (t == 0) out[0] = -v;
}

extern "C" void kernel_launch(void* const* d_in, const int* in_sizes, int n_in,
                              void* d_out, int out_size, void* d_ws, size_t ws_size,
                              hipStream_t stream) {
  const float* pred_bboxes = (const float*)d_in[0];
  const float* pred_labels = (const float*)d_in[1];
  const float* gt_bboxes   = (const float*)d_in[2];
  const int*   gt_labels   = (const int*)d_in[3];
  float* out = (float*)d_out;

  // workspace carve-up (256B aligned)
  auto align256 = [](uintptr_t p) { return (p + 255) & ~(uintptr_t)255; };
  uintptr_t p = (uintptr_t)d_ws;
  float*    palphaC = (float*)p;    p = align256(p + sizeof(float)    * (size_t)BS * NC * KP);
  _Float16* palphaH = (_Float16*)p; p = align256(p + sizeof(_Float16) * (size_t)BS * KP_PAD * K_PAD);
  _Float16* palphaT = (_Float16*)p; p = align256(p + sizeof(_Float16) * (size_t)BS * K_PAD * KP_PAD);
  float*    pstat   = (float*)p;    p = align256(p + sizeof(float)    * (size_t)BS * KP_PAD * SSTR);
  float*    gstat   = (float*)p;    p = align256(p + sizeof(float)    * (size_t)BS * KG * SSTR);
  float*    pq      = (float*)p;    p = align256(p + sizeof(float)    * BS);
  float*    qq      = (float*)p;    p = align256(p + sizeof(float)    * BS);
  float*    pp      = (float*)p;    p = align256(p + sizeof(float)    * BS);
  (void)ws_size; (void)n_in; (void)in_sizes; (void)out_size;

  init_acc<<<1, 64, 0, stream>>>(pq, qq);
  pred_prep<<<BS * KP_PAD, 32, 0, stream>>>(pred_bboxes, pred_labels,
                                            palphaC, palphaH, palphaT, pstat);
  gt_prep<<<(BS * KG + 255) / 256, 256, 0, stream>>>(gt_bboxes, gstat, BS * KG);
  pq_kernel<<<BS * KG, 256, 0, stream>>>(palphaC, pstat, gstat, gt_labels, pq);
  pp_kernel<<<BS, 256, 0, stream>>>(gstat, gt_labels, pp);
  qq_wmma<<<(BS * NTIL * NTIL) / 4, 128, 0, stream>>>(palphaH, palphaT, pstat, qq);
  final_kernel<<<1, 32, 0, stream>>>(pq, pp, qq, out);
}
